// MultiHeadAttention_19370302505368
// MI455X (gfx1250) — matrix operand order repaired and checked by reading, not compiled, not run on hardware
//
#include <hip/hip_runtime.h>
#include <hip/hip_bf16.h>

// MultiHeadAttention: B=4, S=2048, H=1024, NH=16, HD=64, scale = 1/sqrt(H) = 1/32
#define Bc  4
#define Sc  2048
#define Hc  1024
#define NHc 16
#define HDc 64

// scale/sqrt(H) folded with log2(e): softmax runs in base-2 domain
#define SCL2 0.0450842200277980f   // 0.03125 * 1.4426950408889634

typedef __attribute__((ext_vector_type(16))) __bf16 v16bf;
typedef __attribute__((ext_vector_type(8)))  float  v8f;
typedef __attribute__((ext_vector_type(4)))  unsigned int v4u;
typedef __attribute__((ext_vector_type(8)))  int    v8i;
typedef __attribute__((ext_vector_type(4)))  int    v4i;

union FragU { uint4 u[2]; v16bf v; };

__device__ __forceinline__ unsigned short f2bf(float f) {
  unsigned u = __float_as_uint(f);
  u += 0x7FFFu + ((u >> 16) & 1u);   // round-to-nearest-even
  return (unsigned short)(u >> 16);
}

// Truncating f32->bf16: 1 VALU op; used for softmax probabilities in [0,1]
// where the truncation bias is far below bf16 quantization noise.
__device__ __forceinline__ unsigned short f2bf_trunc(float f) {
  return (unsigned short)(__float_as_uint(f) >> 16);
}

// Load a 16-element bf16 fragment as two 16B chunks (both must be 16B aligned).
__device__ __forceinline__ v16bf ldfrag(const unsigned short* p0, const unsigned short* p1) {
  FragU f;
  f.u[0] = *(const uint4*)p0;
  f.u[1] = *(const uint4*)p1;
  return f.v;
}

// CDNA5 async copy: global -> LDS, 16B per lane, tracked by ASYNCcnt.
__device__ __forceinline__ void async_cp_b128(unsigned short* lds_dst,
                                              const unsigned short* g_src) {
  // Generic LDS address is {SHARED_BASE, 32'offset}: low 32 bits = LDS byte addr.
  unsigned ldsa = (unsigned)(uintptr_t)lds_dst;
  asm volatile("global_load_async_to_lds_b128 %0, %1, off"
               :: "v"(ldsa), "v"(g_src) : "memory");
}

// CDNA5 Tensor Data Mover: DMA a 2D bf16 tile (rows x rowlen elems, global row
// stride row_stride elems) into LDS with hardware row padding of 16B per 128B
// (pad_interval=32 DWORDs, pad_amount=4 DWORDs) -> 144B padded LDS rows.
// D# bitfields per CDNA5 ISA 8.3/8.4. Tracked by TENSORcnt.
__device__ __forceinline__ void tdm_load_2d(unsigned lds_addr, const void* gaddr,
                                            unsigned rows, unsigned rowlen,
                                            unsigned row_stride) {
  const unsigned long long ga = (unsigned long long)(uintptr_t)gaddr;
  v4u g0;
  g0.x = 1u;                                            // count=1, user D#
  g0.y = lds_addr;                                      // [63:32] lds_addr
  g0.z = (unsigned)ga;                                  // [95:64] global_addr lo
  g0.w = (unsigned)((ga >> 32) & 0x01FFFFFFu)           // [120:96] global_addr hi
       | (2u << 30);                                    // [127:126] type=2
  v8i g1;
  g1[0] = (int)((1u << 16)                              // data_size = 2 bytes
              | (1u << 20)                              // pad_enable
              | (4u << 22)                              // pad_interval: 32 DW (128B)
              | (3u << 25));                            // pad_amount: 4 DW (16B)
  g1[1] = (int)(rowlen << 16);                          // tensor_dim0[15:0]
  g1[2] = (int)(rows << 16);                            // tensor_dim1[15:0]
  g1[3] = (int)(rowlen << 16);                          // tile_dim0
  g1[4] = (int)rows;                                    // tile_dim1
  g1[5] = (int)row_stride;                              // tensor_dim0_stride[31:0]
  g1[6] = 0;
  g1[7] = 0;
  const v4i z4 = {0, 0, 0, 0};
#if __clang_major__ >= 23
  const v8i z8 = {0, 0, 0, 0, 0, 0, 0, 0};
  __builtin_amdgcn_tensor_load_to_lds(g0, g1, z4, z4, z8, 0);
#else
  __builtin_amdgcn_tensor_load_to_lds(g0, g1, z4, z4, 0);
#endif
}

__global__ void cvt_f32_bf16(const float* __restrict__ in,
                             unsigned short* __restrict__ out, int n) {
  int i = blockIdx.x * blockDim.x + threadIdx.x;
  if (i < n) out[i] = f2bf(in[i]);
}

// ---------------------------------------------------------------------------
// GEMM: Y[M,N] = A[M,K] @ W[N,K]^T + bias,  M=8192, N=K=1024, bf16 in, f32 acc
// Block: 256 thr = 8 waves (2 m-waves x 4 n-waves). Wave tile 32x64 (2x4 WMMA).
// Block tile 64x256; A/B k-slices staged in LDS with double-buffered async
// copies (5 async b128 per wave per tile -> s_wait_asynccnt 5 pipelines).
// MODE 0: store bf16 head-major [b,h,s,d]   (Q, K)
// MODE 1: store bf16 transposed [b,h,d,s]   (V -> Vt, feeds PV B-operand)
// MODE 2: store f32 row-major [M,N]         (output projection)
// ---------------------------------------------------------------------------
template<int MODE>
__global__ void __launch_bounds__(256) gemm_bf16(
    const unsigned short* __restrict__ A,
    const unsigned short* __restrict__ W,
    const float* __restrict__ bias,
    void* __restrict__ outv)
{
  // Padded row stride (40 halves = 80B, 16B aligned) to spread LDS banks.
  __shared__ __align__(16) unsigned short Al[2][64 * 40];
  __shared__ __align__(16) unsigned short Bl[2][256 * 40];

  const int t    = threadIdx.x;
  const int lane = t & 31;
  const int wid  = t >> 5;
  const int lcol = lane & 15;
  const int hl   = lane >> 4;
  const int wm   = wid & 1;
  const int wn   = wid >> 1;
  const int mb   = blockIdx.x * 64;
  const int nb   = blockIdx.y * 256;

  // copy assignment: A tile 64x32 halves = 256 b128; B tile 256x32 = 1024 b128
  const int arow = t >> 2, achk = t & 3;

  auto issueAB = [&](int buf, int k0) {
    async_cp_b128(&Al[buf][arow * 40 + achk * 8],
                  A + (size_t)(mb + arow) * Hc + k0 + achk * 8);
#pragma unroll
    for (int j = 0; j < 4; j++) {
      const int i = t + 256 * j;
      const int brow = i >> 2, bchk = i & 3;
      async_cp_b128(&Bl[buf][brow * 40 + bchk * 8],
                    W + (size_t)(nb + brow) * Hc + k0 + bchk * 8);
    }
  };

  v8f acc[2][4] = {};

  issueAB(0, 0);
  int cur = 0;
  for (int k0 = 0; k0 < Hc; k0 += 32) {
    if (k0 + 32 < Hc) {
      issueAB(cur ^ 1, k0 + 32);
      asm volatile("s_wait_asynccnt 5" ::: "memory");   // current tile landed
    } else {
      asm volatile("s_wait_asynccnt 0" ::: "memory");
    }
    __syncthreads();

    v16bf afr[2], bfr[4];
#pragma unroll
    for (int mf = 0; mf < 2; mf++) {
      const unsigned short* p = &Al[cur][(wm * 32 + mf * 16 + lcol) * 40 + hl * 8];
      afr[mf] = ldfrag(p, p + 16);
    }
#pragma unroll
    for (int nf = 0; nf < 4; nf++) {
      const unsigned short* p = &Bl[cur][(wn * 64 + nf * 16 + lcol) * 40 + hl * 8];
      bfr[nf] = ldfrag(p, p + 16);
    }
#pragma unroll
    for (int mf = 0; mf < 2; mf++)
#pragma unroll
      for (int nf = 0; nf < 4; nf++)
        acc[mf][nf] = __builtin_amdgcn_wmma_f32_16x16x32_bf16(
            false, afr[mf], false, bfr[nf], (short)0, acc[mf][nf], false, false);

    __syncthreads();                                    // before buffer reuse
    cur ^= 1;
  }

  const int mbw = mb + wm * 32;
  const int nbw = nb + wn * 64;
#pragma unroll
  for (int mf = 0; mf < 2; mf++) {
#pragma unroll
    for (int nf = 0; nf < 4; nf++) {
      const int ngl = nbw + nf * 16 + lcol;
      const float bn = bias[ngl];
#pragma unroll
      for (int r = 0; r < 8; r++) {
        const int mgl = mbw + mf * 16 + r + 8 * hl;  // C layout: row = r + 8*half
        const float v = acc[mf][nf][r] + bn;
        if (MODE == 2) {
          ((float*)outv)[(size_t)mgl * Hc + ngl] = v;
        } else {
          const int bi = mgl >> 11, s = mgl & (Sc - 1);
          const int h  = ngl >> 6,  d = ngl & (HDc - 1);
          size_t addr;
          if (MODE == 0) addr = (((size_t)(bi * NHc + h)) * Sc + s) * HDc + d;
          else           addr = (((size_t)(bi * NHc + h)) * HDc + d) * Sc + s;
          ((unsigned short*)outv)[addr] = f2bf(v);
        }
      }
    }
  }
}

// ---------------------------------------------------------------------------
// Flash attention: 8 waves/block, one 16-row q-tile per wave, all waves share
// one (b,h). K and Vt 64-key tiles are DMA'd to LDS by the Tensor Data Mover
// (one tensor_load_to_lds per tile, issued by wave 0; hardware padding gives
// the 144B LDS row stride), double buffered and drained with
// s_wait_tensorcnt. Softmax runs in exp2 domain; the row-sum l is accumulated
// by an extra PV n-tile whose V columns are [1,0,...,0] (constant rows 64..79
// of the V LDS tile), so it inherits the alpha rescale from the WMMA
// accumulator and no cross-lane sum reduction is needed.
// ---------------------------------------------------------------------------
__global__ void __launch_bounds__(256) attn_fa(
    const unsigned short* __restrict__ Q,
    const unsigned short* __restrict__ K,
    const unsigned short* __restrict__ Vt,
    const float* __restrict__ masks,
    unsigned short* __restrict__ ctx)
{
  // 72-half (144B) padded row stride, 16B aligned; matches TDM pad output.
  __shared__ __align__(16) unsigned short Kl[2][64 * 72];  // [key][d]
  __shared__ __align__(16) unsigned short Vl[2][80 * 72];  // [d][key], rows 64..79 = ones-column tile
  __shared__ __align__(16) unsigned short Plds[8][16][64];

  const int t    = threadIdx.x;
  const int lane = t & 31;
  const int wid  = t >> 5;
  const int lcol = lane & 15;
  const int hl   = lane >> 4;
  const int bh   = blockIdx.x >> 4;         // (b*NH + h)
  const int qblk = blockIdx.x & 15;         // 16 q-blocks of 128 rows each
  const int bi   = bh >> 4;
  const int h    = bh & (NHc - 1);
  const int qb   = qblk * 128 + wid * 16;

  const unsigned short* Qh = Q  + (size_t)bh * Sc * HDc;
  const unsigned short* Kh = K  + (size_t)bh * Sc * HDc;
  const unsigned short* Vh = Vt + (size_t)bh * HDc * Sc;
  const float* mk = masks + (size_t)bi * Sc;

  // Wave 0 drives the TDM: K tile = 64 key-rows x 64 d; V tile = 64 d-rows x
  // 64 keys (row stride S). TENSORcnt is wave-private; the block barrier after
  // wave 0's s_wait_tensorcnt publishes the tiles to all waves.
  auto issueKV = [&](int buf, int kt) {
    if (wid == 0) {
      tdm_load_2d((unsigned)(uintptr_t)&Kl[buf][0], Kh + (size_t)kt * HDc,
                  64, 64, HDc);
      tdm_load_2d((unsigned)(uintptr_t)&Vl[buf][0], Vh + kt,
                  64, 64, Sc);
    }
  };

  // Constant rows 64..79 of both V buffers: dim 64 = 1.0 (l-accumulator
  // column), dims 65..79 = 0. Written once; first in-loop barrier publishes.
  for (int i = t; i < 2 * 16 * 64; i += 256) {
    const int buf = i >> 10;
    const int rr  = (i >> 6) & 15;
    const int cc  = i & 63;
    Vl[buf][(64 + rr) * 72 + cc] =
        (rr == 0) ? (unsigned short)0x3F80 : (unsigned short)0;
  }

  // Q fragments stay in registers for the whole key loop (hd=64 -> 2 k-steps)
  v16bf qf[2];
#pragma unroll
  for (int ks = 0; ks < 2; ks++) {
    const unsigned short* p = Qh + (size_t)(qb + lcol) * HDc + ks * 32 + hl * 8;
    qf[ks] = ldfrag(p, p + 16);
  }

  float mrun[8];
  v8f acc[5] = {};                          // [0..3]=ctx dims, [4] col0 = l
#pragma unroll
  for (int r = 0; r < 8; r++) mrun[r] = -3.0e38f;

  issueKV(0, 0);
  int cur = 0;
  for (int kt = 0; kt < Sc; kt += 64) {
    if (kt + 64 < Sc) {
      issueKV(cur ^ 1, kt + 64);
      if (wid == 0) __builtin_amdgcn_s_wait_tensorcnt((short)2);  // cur landed
    } else {
      if (wid == 0) __builtin_amdgcn_s_wait_tensorcnt((short)0);
    }
    __syncthreads();

    // scores: load all 8 K fragments first, then chain 8 WMMAs
    v8f sc[4] = {};
    {
      v16bf kf[8];
#pragma unroll
      for (int nt = 0; nt < 4; nt++)
#pragma unroll
        for (int ks = 0; ks < 2; ks++) {
          const unsigned short* p =
              &Kl[cur][(nt * 16 + lcol) * 72 + ks * 32 + hl * 8];
          kf[nt * 2 + ks] = ldfrag(p, p + 16);
        }
#pragma unroll
      for (int ks = 0; ks < 2; ks++)
#pragma unroll
        for (int nt = 0; nt < 4; nt++)
          sc[nt] = __builtin_amdgcn_wmma_f32_16x16x32_bf16(
              false, qf[ks], false, kf[nt * 2 + ks], (short)0, sc[nt], false, false);
    }

    // mask folded into an additive bias (0 or -1e12), combined with the
    // scale*log2e factor in a single FMA per element.
    float mbias[4];
#pragma unroll
    for (int nt = 0; nt < 4; nt++)
      mbias[nt] = (mk[kt + nt * 16 + lcol] != 0.f) ? 0.f : -1.0e12f;

    // online softmax, base-2 domain; row r (+8*hl) spread over 16 lanes
    float p4[4][8];
#pragma unroll
    for (int r = 0; r < 8; r++) {
      float sv[4];
      float tmx = -3.0e38f;
#pragma unroll
      for (int nt = 0; nt < 4; nt++) {
        sv[nt] = fmaf(sc[nt][r], SCL2, mbias[nt]);
        tmx = fmaxf(tmx, sv[nt]);
      }
#pragma unroll
      for (int off = 8; off >= 1; off >>= 1)
        tmx = fmaxf(tmx, __shfl_xor(tmx, off, 32));   // max over 16-lane half
      const float nm = fmaxf(mrun[r], tmx);
      const float alpha = __builtin_amdgcn_exp2f(mrun[r] - nm);
      mrun[r] = nm;
#pragma unroll
      for (int nt = 0; nt < 4; nt++)
        p4[nt][r] = __builtin_amdgcn_exp2f(sv[nt] - nm);
#pragma unroll
      for (int j = 0; j < 5; j++) acc[j][r] = acc[j][r] * alpha;
    }

    // stage P (16x64) into this wave's LDS slice, C-layout -> row-major bf16
    // (truncating conversion: 1 VALU op per element)
#pragma unroll
    for (int nt = 0; nt < 4; nt++)
#pragma unroll
      for (int r = 0; r < 8; r++)
        Plds[wid][r + 8 * hl][nt * 16 + lcol] = f2bf_trunc(p4[nt][r]);

    // P @ V : 2 k-steps x 5 n-tiles (4 ctx + 1 l-accumulator)
#pragma unroll
    for (int kk = 0; kk < 2; kk++) {
      const unsigned short* pp = &Plds[wid][lcol][kk * 32 + hl * 8];
      v16bf pf = ldfrag(pp, pp + 16);
      v16bf vf[5];
#pragma unroll
      for (int nt = 0; nt < 5; nt++) {
        const unsigned short* vp =
            &Vl[cur][(nt * 16 + lcol) * 72 + kk * 32 + hl * 8];
        vf[nt] = ldfrag(vp, vp + 16);
      }
#pragma unroll
      for (int nt = 0; nt < 5; nt++)
        acc[nt] = __builtin_amdgcn_wmma_f32_16x16x32_bf16(
            false, pf, false, vf[nt], (short)0, acc[nt], false, false);
    }

    __syncthreads();                      // before K/V buffer reuse
    cur ^= 1;
  }

  // normalize: l for row (r, hl) sits at lane hl*16, col 0 of acc[4]
#pragma unroll
  for (int r = 0; r < 8; r++) {
    const float lv = __shfl(acc[4][r], hl << 4, 32);
    const float inv = lv > 0.f ? __builtin_amdgcn_rcpf(lv) : 0.f;
    const int s = qb + r + 8 * hl;
#pragma unroll
    for (int nt = 0; nt < 4; nt++)
      ctx[((size_t)(bi * Sc + s)) * Hc + h * HDc + nt * 16 + lcol] =
          f2bf(acc[nt][r] * inv);
  }
}

extern "C" void kernel_launch(void* const* d_in, const int* in_sizes, int n_in,
                              void* d_out, int out_size, void* d_ws, size_t ws_size,
                              hipStream_t stream) {
  (void)in_sizes; (void)n_in; (void)out_size; (void)ws_size;
  const float* hs = (const float*)d_in[0];
  const float* mk = (const float*)d_in[1];
  const float* Wq = (const float*)d_in[2];
  const float* bq = (const float*)d_in[3];
  const float* Wk = (const float*)d_in[4];
  const float* bk = (const float*)d_in[5];
  const float* Wv = (const float*)d_in[6];
  const float* bv = (const float*)d_in[7];
  const float* Wo = (const float*)d_in[8];
  const float* bo = (const float*)d_in[9];

  char* ws = (char*)d_ws;
  const size_t XB = (size_t)Bc * Sc * Hc * sizeof(unsigned short);  // 16 MiB
  const size_t WB = (size_t)Hc * Hc * sizeof(unsigned short);       //  2 MiB
  unsigned short* Xb  = (unsigned short*)(ws);
  unsigned short* Wqb = (unsigned short*)(ws + XB);
  unsigned short* Wkb = (unsigned short*)(ws + XB + 1 * WB);
  unsigned short* Wvb = (unsigned short*)(ws + XB + 2 * WB);
  unsigned short* Wob = (unsigned short*)(ws + XB + 3 * WB);
  unsigned short* Qb  = (unsigned short*)(ws + 1 * XB + 4 * WB);
  unsigned short* Kb  = (unsigned short*)(ws + 2 * XB + 4 * WB);
  unsigned short* Vtb = (unsigned short*)(ws + 3 * XB + 4 * WB);
  unsigned short* Cxb = (unsigned short*)(ws + 4 * XB + 4 * WB);

  const int NX = Bc * Sc * Hc;   // 8388608
  const int NW = Hc * Hc;        // 1048576
  cvt_f32_bf16<<<(NX + 255) / 256, 256, 0, stream>>>(hs, Xb, NX);
  cvt_f32_bf16<<<(NW + 255) / 256, 256, 0, stream>>>(Wq, Wqb, NW);
  cvt_f32_bf16<<<(NW + 255) / 256, 256, 0, stream>>>(Wk, Wkb, NW);
  cvt_f32_bf16<<<(NW + 255) / 256, 256, 0, stream>>>(Wv, Wvb, NW);
  cvt_f32_bf16<<<(NW + 255) / 256, 256, 0, stream>>>(Wo, Wob, NW);

  dim3 gg(Bc * Sc / 64, Hc / 256);   // 128 x 4
  gemm_bf16<0><<<gg, 256, 0, stream>>>(Xb, Wqb, bq, Qb);
  gemm_bf16<0><<<gg, 256, 0, stream>>>(Xb, Wkb, bk, Kb);
  gemm_bf16<1><<<gg, 256, 0, stream>>>(Xb, Wvb, bv, Vtb);

  attn_fa<<<Bc * NHc * (Sc / 128), 256, 0, stream>>>(Qb, Kb, Vtb, mk, Cxb);

  gemm_bf16<2><<<gg, 256, 0, stream>>>(Cxb, Wob, bo, d_out);
}
